// MultiHeadAttention_57200374448535
// MI455X (gfx1250) — matrix operand order repaired and checked by reading, not compiled, not run on hardware
//
#include <hip/hip_runtime.h>
#include <hip/hip_bf16.h>
#include <math.h>

typedef __attribute__((ext_vector_type(16))) _Float16 v16h;
typedef __attribute__((ext_vector_type(8)))  float    v8f;

#define NEGINF (-__builtin_inff())

// ---------------------------------------------------------------------------
// Fragment loaders (layouts per CDNA5 ISA 7.12.2, wave32)
// ---------------------------------------------------------------------------

// A-matrix 16x32 f16: lane (l<16) -> row l, k = {0..7, 16..23}; lane>=16 -> row l-16, k = {8..15, 24..31}
template <typename AT>
__device__ __forceinline__ v16h load_a_frag(const AT* __restrict__ A, int lda,
                                            int m0, int k0, int lane) {
    int m  = m0 + (lane & 15);
    int kb = k0 + ((lane >> 4) << 3);
    const AT* p = A + (long)m * lda + kb;
    v16h a;
#pragma unroll
    for (int i = 0; i < 8; ++i) {
        a[i]     = (_Float16)p[i];
        a[i + 8] = (_Float16)p[i + 16];
    }
    return a;
}

// B-matrix 32x16 f16 from B^T stored row-major [N][K]:
// lane -> column n = lane&15 ; lanes 0-15 k = {0..7, 16..23}, lanes 16-31 k = {8..15, 24..31} (relative to k0)
__device__ __forceinline__ v16h load_b_frag(const _Float16* __restrict__ Bt, long ldb,
                                            int n0, int k0, int lane) {
    int n  = n0 + (lane & 15);
    int kb = k0 + ((lane >> 4) << 3);
    const _Float16* p = Bt + (long)n * ldb + kb;
    v16h b;
#pragma unroll
    for (int i = 0; i < 8; ++i) {
        b[i]     = p[i];
        b[i + 8] = p[i + 16];
    }
    return b;
}

__device__ __forceinline__ v8f wmma_f16(v16h a, v16h b, v8f c) {
    return __builtin_amdgcn_wmma_f32_16x16x32_f16(false, a, false, b, (short)0, c,
                                                  false, false);
}

__device__ __forceinline__ float half_reduce_max(float v) {
    v = fmaxf(v, __shfl_xor(v, 1, 32));
    v = fmaxf(v, __shfl_xor(v, 2, 32));
    v = fmaxf(v, __shfl_xor(v, 4, 32));
    v = fmaxf(v, __shfl_xor(v, 8, 32));
    return v;
}
__device__ __forceinline__ float half_reduce_sum(float v) {
    v += __shfl_xor(v, 1, 32);
    v += __shfl_xor(v, 2, 32);
    v += __shfl_xor(v, 4, 32);
    v += __shfl_xor(v, 8, 32);
    return v;
}

// ---------------------------------------------------------------------------
// fp32 -> f16 conversion (weights)
// ---------------------------------------------------------------------------
__global__ void cvt_f32_f16(const float* __restrict__ s, _Float16* __restrict__ d, int n) {
    int i = (blockIdx.x * blockDim.x + threadIdx.x) * 4;
    if (i + 3 < n) {
        float4 v = *(const float4*)(s + i);
        d[i + 0] = (_Float16)v.x;
        d[i + 1] = (_Float16)v.y;
        d[i + 2] = (_Float16)v.z;
        d[i + 3] = (_Float16)v.w;
    }
}

// ---------------------------------------------------------------------------
// WMMA GEMM:  C[M,N] = alpha * A[M,K] * Bt[N,K]^T   (x @ W.T form)
// MODE 0: f16 out, row-major         (Q with alpha=0.125, K with alpha=1)
// MODE 1: f16 out, per-head V^T:     dst[(bi*N + n)*2048 + (m&2047)]
// MODE 2: f32 out + bias             (final projection)
// Block: 256 threads = 8 waves, block tile 64x32, one 16x16 WMMA tile / wave.
// ---------------------------------------------------------------------------
template <typename AT, int MODE>
__global__ __launch_bounds__(256) void gemm_wmma(const AT* __restrict__ A,
                                                 const _Float16* __restrict__ Bt,
                                                 void* __restrict__ Cout,
                                                 const float* __restrict__ bias,
                                                 int M, int N, int K, float alpha) {
    int lane = threadIdx.x & 31;
    int w    = threadIdx.x >> 5;                 // 0..7
    int m0   = blockIdx.y * 64 + (w & 3) * 16;
    int n0   = blockIdx.x * 32 + (w >> 2) * 16;

    v8f c = {};
    for (int k0 = 0; k0 < K; k0 += 32) {
        if (k0 + 64 < K)
            __builtin_prefetch(A + (long)(m0 + (lane & 15)) * K + k0 + 64, 0, 1);
        v16h a = load_a_frag<AT>(A, K, m0, k0, lane);
        v16h b = load_b_frag(Bt, K, n0, k0, lane);
        c = wmma_f16(a, b, c);
    }

    int n  = lane & 15;
    int mb = (lane >> 4) * 8;
    if (MODE == 0) {
        _Float16* D = (_Float16*)Cout;
#pragma unroll
        for (int r = 0; r < 8; ++r)
            D[(long)(m0 + mb + r) * N + (n0 + n)] = (_Float16)(alpha * c[r]);
    } else if (MODE == 1) {
        _Float16* D = (_Float16*)Cout;
#pragma unroll
        for (int r = 0; r < 8; ++r) {
            int m  = m0 + mb + r;
            int bi = m >> 11;  // batch index (L = 2048)
            D[((long)(bi * N + (n0 + n))) * 2048 + (m & 2047)] = (_Float16)c[r];
        }
    } else {
        float* D = (float*)Cout;
        float  bv = bias[n0 + n];
#pragma unroll
        for (int r = 0; r < 8; ++r)
            D[(long)(m0 + mb + r) * N + (n0 + n)] = c[r] + bv;
    }
}

// ---------------------------------------------------------------------------
// Flash attention (scale folded into Q; key padding mask; NaN rows -> 0)
// Q,K: f16 (B*L, 1024) ; Vt: f16 (B,H,64,L) ; Ctx out: f16 (B*L, 1024)
// Block: 128 threads = 4 waves, each wave owns 16 query rows; key tiles of 32.
// ---------------------------------------------------------------------------
__global__ __launch_bounds__(128) void attn_wmma(const _Float16* __restrict__ Qh,
                                                 const _Float16* __restrict__ Kh,
                                                 const _Float16* __restrict__ Vt,
                                                 const int* __restrict__ maskI,
                                                 _Float16* __restrict__ Ctx) {
    constexpr int L = 2048, HD = 64, D = 1024;
    __shared__ _Float16 plds[4][16][36];  // per-wave P staging (C-layout -> A-layout)

    int lane = threadIdx.x & 31;
    int w    = threadIdx.x >> 5;
    int bh   = blockIdx.x;       // b*16 + h
    int b    = bh >> 4;
    int h    = bh & 15;
    int m0   = blockIdx.y * 64 + w * 16;   // query rows for this wave

    const _Float16* Qbase = Qh + (long)(b * L) * D + h * HD;
    const _Float16* Kbase = Kh + (long)(b * L) * D + h * HD;
    const _Float16* Vtb   = Vt + (long)bh * HD * L;
    const int*      mrow  = maskI + b * L;

    // Q fragments for the whole pass (hd = 64 -> two K=32 chunks)
    v16h q0 = load_a_frag<_Float16>(Qbase, D, m0, 0, lane);
    v16h q1 = load_a_frag<_Float16>(Qbase, D, m0, 32, lane);

    v8f o[4];
#pragma unroll
    for (int j = 0; j < 4; ++j) o[j] = (v8f){};

    float m_r[8], l_r[8];
#pragma unroll
    for (int r = 0; r < 8; ++r) { m_r[r] = NEGINF; l_r[r] = 0.0f; }

    int colkey = lane & 15;

    for (int kt = 0; kt < L; kt += 32) {
        // ---- S = Q K^T (scale pre-folded into Q) : two 16-key column tiles
        v8f s0 = (v8f){}, s1 = (v8f){};
        s0 = wmma_f16(q0, load_b_frag(Kbase, D, kt, 0, lane), s0);
        s0 = wmma_f16(q1, load_b_frag(Kbase, D, kt, 32, lane), s0);
        s1 = wmma_f16(q0, load_b_frag(Kbase, D, kt + 16, 0, lane), s1);
        s1 = wmma_f16(q1, load_b_frag(Kbase, D, kt + 16, 32, lane), s1);

        bool keep0 = mrow[kt + colkey] != 0;
        bool keep1 = mrow[kt + 16 + colkey] != 0;

        float corr[8];
#pragma unroll
        for (int r = 0; r < 8; ++r) {
            float a0 = keep0 ? s0[r] : NEGINF;
            float a1 = keep1 ? s1[r] : NEGINF;
            float t  = half_reduce_max(fmaxf(a0, a1));
            float mo = m_r[r];
            float mn = fmaxf(mo, t);
            float cf, p0, p1;
            if (mn == NEGINF) {           // tile (and history) fully masked
                cf = 1.0f; p0 = 0.0f; p1 = 0.0f;
            } else {
                cf = (mo == NEGINF) ? 0.0f : __expf(mo - mn);
                p0 = (a0 == NEGINF) ? 0.0f : __expf(a0 - mn);
                p1 = (a1 == NEGINF) ? 0.0f : __expf(a1 - mn);
            }
            float rs = half_reduce_sum(p0 + p1);
            l_r[r]   = l_r[r] * cf + rs;
            m_r[r]   = mn;
            corr[r]  = cf;
            s0[r]    = p0;  // reuse as P
            s1[r]    = p1;
        }
#pragma unroll
        for (int j = 0; j < 4; ++j)
#pragma unroll
            for (int r = 0; r < 8; ++r) o[j][r] *= corr[r];

        // ---- P: C-layout -> A-fragment via per-wave LDS bounce
        __builtin_amdgcn_wave_barrier();
#pragma unroll
        for (int r = 0; r < 8; ++r) {
            int row = r + (lane >> 4) * 8;
            plds[w][row][colkey]      = (_Float16)s0[r];
            plds[w][row][colkey + 16] = (_Float16)s1[r];
        }
        __builtin_amdgcn_wave_barrier();
        asm volatile("s_wait_dscnt 0x0" ::: "memory");
        v16h pa;
        {
            int m  = lane & 15;
            int kb = (lane >> 4) << 3;
#pragma unroll
            for (int i = 0; i < 8; ++i) {
                pa[i]     = plds[w][m][kb + i];
                pa[i + 8] = plds[w][m][kb + 16 + i];
            }
        }
        __builtin_amdgcn_wave_barrier();

        // ---- O += P * V   (V^T layout -> contiguous B fragments)
#pragma unroll
        for (int j = 0; j < 4; ++j)
            o[j] = wmma_f16(pa, load_b_frag(Vtb, L, j * 16, kt, lane), o[j]);
    }

    // ---- finalize: divide by l; fully-masked rows -> 0 (matches NaN zeroing)
    float sc[8];
#pragma unroll
    for (int r = 0; r < 8; ++r)
        sc[r] = (m_r[r] == NEGINF || l_r[r] <= 0.0f) ? 0.0f : 1.0f / l_r[r];

    int mb = (lane >> 4) * 8;
#pragma unroll
    for (int j = 0; j < 4; ++j)
#pragma unroll
        for (int r = 0; r < 8; ++r) {
            long row = (long)(b * L + m0 + mb + r);
            Ctx[row * D + h * HD + j * 16 + colkey] = (_Float16)(o[j][r] * sc[r]);
        }
}

// ---------------------------------------------------------------------------
// Host launcher
// ---------------------------------------------------------------------------
extern "C" void kernel_launch(void* const* d_in, const int* in_sizes, int n_in,
                              void* d_out, int out_size, void* d_ws, size_t ws_size,
                              hipStream_t stream) {
    const float* query = (const float*)d_in[0];
    const float* key   = (const float*)d_in[1];
    const float* value = (const float*)d_in[2];
    const int*   mask  = (const int*)d_in[3];
    const float* Wq    = (const float*)d_in[4];
    const float* Wk    = (const float*)d_in[5];
    const float* Wv    = (const float*)d_in[6];
    const float* Wo    = (const float*)d_in[7];
    const float* bo    = (const float*)d_in[8];
    float*       out   = (float*)d_out;

    constexpr int B = 2, L = 2048, D = 1024;
    constexpr int M = B * L;                     // 4096
    const size_t wBytes = (size_t)D * D * 2;     // 2 MB per f16 weight
    const size_t aBytes = (size_t)M * D * 2;     // 8 MB per f16 activation

    char* ws = (char*)d_ws;
    _Float16* Wq_h  = (_Float16*)(ws);                       // 2 MB
    _Float16* Wk_h  = (_Float16*)(ws + wBytes);              // 2 MB
    _Float16* Wv_h  = (_Float16*)(ws + 2 * wBytes);          // 2 MB
    _Float16* Wo_h  = (_Float16*)(ws + 3 * wBytes);          // 2 MB
    _Float16* Q_h   = (_Float16*)(ws + 4 * wBytes);          // 8 MB
    _Float16* K_h   = (_Float16*)(ws + 4 * wBytes + aBytes); // 8 MB
    _Float16* Vt_h  = (_Float16*)(ws + 4 * wBytes + 2 * aBytes);
    _Float16* Ctx_h = (_Float16*)(ws + 4 * wBytes + 3 * aBytes);

    // weights fp32 -> f16
    int wn = D * D;
    dim3 cg(wn / (256 * 4)), cb(256);
    cvt_f32_f16<<<cg, cb, 0, stream>>>(Wq, Wq_h, wn);
    cvt_f32_f16<<<cg, cb, 0, stream>>>(Wk, Wk_h, wn);
    cvt_f32_f16<<<cg, cb, 0, stream>>>(Wv, Wv_h, wn);
    cvt_f32_f16<<<cg, cb, 0, stream>>>(Wo, Wo_h, wn);

    // projections (WMMA). Q carries the 1/sqrt(64) softmax scale.
    dim3 gp(D / 32, M / 64), bp(256);
    gemm_wmma<float, 0><<<gp, bp, 0, stream>>>(query, Wq_h, Q_h,  nullptr, M, D, D, 0.125f);
    gemm_wmma<float, 0><<<gp, bp, 0, stream>>>(key,   Wk_h, K_h,  nullptr, M, D, D, 1.0f);
    gemm_wmma<float, 1><<<gp, bp, 0, stream>>>(value, Wv_h, Vt_h, nullptr, M, D, D, 1.0f);

    // fused flash attention
    attn_wmma<<<dim3(B * 16, L / 64), 128, 0, stream>>>(Q_h, K_h, Vt_h, mask, Ctx_h);

    // output projection + bias (fp32 out)
    gemm_wmma<_Float16, 2><<<gp, bp, 0, stream>>>(Ctx_h, Wo_h, out, bo, M, D, D, 1.0f);
}
